// relative_positioning_20091857011045
// MI455X (gfx1250) — matrix operand order repaired and checked by reading, not compiled, not run on hardware
//
#include <hip/hip_runtime.h>
#include <hip/hip_bf16.h>
#include <math.h>

#define N_NODES 4096
#define N_EDGES 16384
#define F_IN    64
#define F_EDGE  16
#define HID     128
#define OUTF    64
#define NGR     32
#define SI      (N_EDGES + N_NODES)

typedef __attribute__((ext_vector_type(16))) __bf16 v16bf;
typedef __attribute__((ext_vector_type(8)))  float  v8f;

__device__ __forceinline__ v8f zero8() {
  v8f c = {0.f, 0.f, 0.f, 0.f, 0.f, 0.f, 0.f, 0.f};
  return c;
}

__device__ __forceinline__ v8f wmma_bf16(v16bf a, v16bf b, v8f c) {
  // D = A(16x32 bf16) * B(32x16 bf16) + C(16x16 f32)
  return __builtin_amdgcn_wmma_f32_16x16x32_bf16(false, a, false, b, (short)0, c,
                                                 false, false);
}

// ---- WMMA tile loaders (wave32 layouts per CDNA5 ISA 7.12.2) ----
// A (16-bit, 16x32 MxK): lane L: M = L%16; element t -> K = (t/8)*16 + (L/16)*8 + t%8

// Full tile, no bounds guard (caller guarantees k0..k0+31 valid)
__device__ __forceinline__ v16bf load_a_full(const float* __restrict__ A, int lda,
                                             int row0, int k0) {
  int lane = threadIdx.x & 31, m = lane & 15, half = lane >> 4;
  const float* rp = A + (size_t)(row0 + m) * lda + k0;
  v16bf a;
#pragma unroll
  for (int t = 0; t < 16; ++t) {
    int k = ((t >> 3) << 4) + (half << 3) + (t & 7);
    a[t] = (__bf16)rp[k];
  }
  return a;
}

// Padded tile: branchless (clamped address + select), no exec divergence
__device__ __forceinline__ v16bf load_a_pad(const float* __restrict__ A, int lda,
                                            int row0, int k0, int kmax) {
  int lane = threadIdx.x & 31, m = lane & 15, half = lane >> 4;
  const float* rp = A + (size_t)(row0 + m) * lda;
  v16bf a;
#pragma unroll
  for (int t = 0; t < 16; ++t) {
    int k = k0 + ((t >> 3) << 4) + (half << 3) + (t & 7);
    int kc = k < kmax ? k : 0;
    float v = rp[kc];
    a[t] = (__bf16)(k < kmax ? v : 0.f);
  }
  return a;
}

__device__ __forceinline__ v16bf load_a_lds(const float* S, int lds, int k0) {
  int lane = threadIdx.x & 31, m = lane & 15, half = lane >> 4;
  v16bf a;
#pragma unroll
  for (int t = 0; t < 16; ++t) {
    int k = k0 + ((t >> 3) << 4) + (half << 3) + (t & 7);
    a[t] = (__bf16)S[m * lds + k];
  }
  return a;
}

// Pre-packed B tile: one v16bf (32B) vector load per lane
__device__ __forceinline__ v16bf load_b_packed(const __bf16* __restrict__ p, int tileIdx) {
  return ((const v16bf*)p)[tileIdx * 32 + (threadIdx.x & 31)];
}

// ---- generic fill ----
__global__ void fill_kernel(float* __restrict__ p, float v, int n) {
  int tid = blockIdx.x * blockDim.x + threadIdx.x;
  if (tid < n) p[tid] = v;
}

// ---- pre-pack e_w3 (f32 [64,8192]) into bf16 WMMA-B layout tiles ----
// tile index ((i*8+nb)*2+ks), within tile: lane*16 + t (32B/lane)
__global__ void pack_w3_kernel(const float* __restrict__ w3, __bf16* __restrict__ w3p) {
  int tid = blockIdx.x * blockDim.x + threadIdx.x;  // 524288 total
  int t    = tid & 15;
  int lane = (tid >> 4) & 31;
  int ks   = (tid >> 9) & 1;
  int nb   = (tid >> 10) & 7;
  int i    = tid >> 13;
  int k    = ks * 32 + ((t >> 3) << 4) + ((lane >> 4) << 3) + (t & 7);
  int col  = i * HID + nb * 16 + (lane & 15);
  w3p[tid] = (__bf16)w3[(size_t)k * (F_IN * HID) + col];
}

// ---- generic pre-pack of a row-major f32 [K,N] weight into bf16 B-layout ----
// tiles laid out (kb * nnb + nbb); K zero-padded to 32*nkb
__global__ void pack_b_kernel(const float* __restrict__ B, __bf16* __restrict__ out,
                              int K, int N, int nkb, int nnb) {
  int tid = blockIdx.x * blockDim.x + threadIdx.x;
  if (tid >= nkb * nnb * 512) return;
  int t    = tid & 15;
  int lane = (tid >> 4) & 31;
  int tileIdx = tid >> 9;
  int nbb = tileIdx % nnb;
  int kb  = tileIdx / nnb;
  int k   = kb * 32 + ((t >> 3) << 4) + ((lane >> 4) << 3) + (t & 7);
  int n   = nbb * 16 + (lane & 15);
  int kc  = k < K ? k : 0;
  float v = B[(size_t)kc * N + n];
  out[tid] = (__bf16)(k < K ? v : 0.f);
}

// ---- K1: edge MLP  ea[E,16] -> relu(@w1+b1)[E,128] -> relu(@w2+b2)[E,64] ----
__global__ void edge_mlp_kernel(const float* __restrict__ ea,
                                const __bf16* __restrict__ w1p, const float* __restrict__ b1p,
                                const __bf16* __restrict__ w2p, const float* __restrict__ b2p,
                                float* __restrict__ h2e) {
  __shared__ float l1[16 * HID];
  int row0 = blockIdx.x * 16;
  int lane = threadIdx.x & 31, n = lane & 15, half = lane >> 4;
  v16bf a = load_a_pad(ea, F_EDGE, row0, 0, F_EDGE);  // K padded 16->32
#pragma unroll
  for (int nb = 0; nb < 8; ++nb) {
    v8f c = zero8();
    c = wmma_bf16(a, load_b_packed(w1p, nb), c);
    float bias = b1p[nb * 16 + n];
#pragma unroll
    for (int r = 0; r < 8; ++r) {
      float v = c[r] + bias;
      l1[(half * 8 + r) * HID + nb * 16 + n] = v > 0.f ? v : 0.f;
    }
  }
  __syncthreads();
  v16bf a2[4];
#pragma unroll
  for (int ks = 0; ks < 4; ++ks) a2[ks] = load_a_lds(l1, HID, ks * 32);
#pragma unroll
  for (int nb = 0; nb < 4; ++nb) {
    v8f c = zero8();
#pragma unroll
    for (int ks = 0; ks < 4; ++ks) c = wmma_bf16(a2[ks], load_b_packed(w2p, ks * 4 + nb), c);
    float bias = b2p[nb * 16 + n];
#pragma unroll
    for (int r = 0; r < 8; ++r) {
      float v = c[r] + bias;
      h2e[(size_t)(row0 + half * 8 + r) * 64 + nb * 16 + n] = v > 0.f ? v : 0.f;
    }
  }
}

// ---- K2 (hot): fused NNConv message generation + scatter-add ----
// msg[e,o] = sum_i x[src[e],i] * ( sum_k h2e[e,k]*W3[k,i*128+o] + b3[i*128+o] )
__global__ void nnconv_kernel(const float* __restrict__ x, const int* __restrict__ src,
                              const int* __restrict__ dst, const float* __restrict__ h2e,
                              const __bf16* __restrict__ w3p, const float* __restrict__ b3,
                              float* __restrict__ agg) {
  __shared__ float xs[16 * F_IN];
  int e0 = blockIdx.x * 16;
  int lane = threadIdx.x & 31, n = lane & 15, half = lane >> 4;
  for (int idx = threadIdx.x; idx < 16 * F_IN; idx += 32) {
    int e = idx >> 6, f = idx & 63;
    xs[idx] = x[(size_t)src[e0 + e] * F_IN + f];
  }
  __syncthreads();
  v16bf a0 = load_a_full(h2e, 64, e0, 0);
  v16bf a1 = load_a_full(h2e, 64, e0, 32);
  v8f macc[8];
#pragma unroll
  for (int nb = 0; nb < 8; ++nb) macc[nb] = zero8();
  for (int i = 0; i < F_IN; ++i) {
    float xr[8];
#pragma unroll
    for (int r = 0; r < 8; ++r) xr[r] = xs[(half * 8 + r) * F_IN + i];
#pragma unroll
    for (int nb = 0; nb < 8; ++nb) {
      int tile0 = (i * 8 + nb) * 2;
      v16bf bb0 = load_b_packed(w3p, tile0);
      v16bf bb1 = load_b_packed(w3p, tile0 + 1);
      v8f t = zero8();
      t = wmma_bf16(a0, bb0, t);
      t = wmma_bf16(a1, bb1, t);
      float bval = b3[i * HID + nb * 16 + n];
#pragma unroll
      for (int r = 0; r < 8; ++r) macc[nb][r] += xr[r] * (t[r] + bval);
    }
  }
#pragma unroll
  for (int nb = 0; nb < 8; ++nb) {
#pragma unroll
    for (int r = 0; r < 8; ++r) {
      int e = e0 + half * 8 + r;
      int d = dst[e];
      atomicAdd(&agg[(size_t)d * HID + nb * 16 + n], macc[nb][r]);
    }
  }
}

// ---- K3: h1 = relu(agg + x@root_w + conv1_b) ----
__global__ void conv_root_kernel(const float* __restrict__ x, const __bf16* __restrict__ rootp,
                                 const float* __restrict__ cb, const float* __restrict__ agg,
                                 float* __restrict__ h1) {
  int row0 = blockIdx.x * 16;
  int lane = threadIdx.x & 31, n = lane & 15, half = lane >> 4;
  v16bf a0 = load_a_full(x, F_IN, row0, 0);
  v16bf a1 = load_a_full(x, F_IN, row0, 32);
#pragma unroll
  for (int nb = 0; nb < 8; ++nb) {
    v8f c = zero8();
    c = wmma_bf16(a0, load_b_packed(rootp, nb), c);
    c = wmma_bf16(a1, load_b_packed(rootp, 8 + nb), c);
    float bias = cb[nb * 16 + n];
#pragma unroll
    for (int r = 0; r < 8; ++r) {
      size_t o = (size_t)(row0 + half * 8 + r) * HID + nb * 16 + n;
      float v = c[r] + agg[o] + bias;
      h1[o] = v > 0.f ? v : 0.f;
    }
  }
}

// ---- K4: hp = h1@gat_w ; a_s = hp@att_src ; a_d = hp@att_dst ----
__global__ void gat_proj_kernel(const float* __restrict__ h1, const __bf16* __restrict__ gatp,
                                const float* __restrict__ att_s, const float* __restrict__ att_d,
                                float* __restrict__ hp, float* __restrict__ a_s,
                                float* __restrict__ a_d) {
  int row0 = blockIdx.x * 16;
  int lane = threadIdx.x & 31, n = lane & 15, half = lane >> 4;
  v16bf a[4];
#pragma unroll
  for (int ks = 0; ks < 4; ++ks) a[ks] = load_a_full(h1, HID, row0, ks * 32);
  float ps[8] = {0.f, 0.f, 0.f, 0.f, 0.f, 0.f, 0.f, 0.f};
  float pd[8] = {0.f, 0.f, 0.f, 0.f, 0.f, 0.f, 0.f, 0.f};
#pragma unroll
  for (int nb = 0; nb < 8; ++nb) {
    v8f c = zero8();
#pragma unroll
    for (int ks = 0; ks < 4; ++ks) c = wmma_bf16(a[ks], load_b_packed(gatp, ks * 8 + nb), c);
    float asv = att_s[nb * 16 + n], adv = att_d[nb * 16 + n];
#pragma unroll
    for (int r = 0; r < 8; ++r) {
      hp[(size_t)(row0 + half * 8 + r) * HID + nb * 16 + n] = c[r];
      ps[r] += c[r] * asv;
      pd[r] += c[r] * adv;
    }
  }
#pragma unroll
  for (int r = 0; r < 8; ++r) {
    float s = ps[r], d = pd[r];
#pragma unroll
    for (int off = 1; off < 16; off <<= 1) {
      s += __shfl_xor(s, off, 32);
      d += __shfl_xor(d, off, 32);
    }
    if (n == 0) {
      a_s[row0 + half * 8 + r] = s;
      a_d[row0 + half * 8 + r] = d;
    }
  }
}

// ---- attention softmax over incoming edges (with self-loops) ----
__device__ __forceinline__ void atomicMaxF(float* addr, float v) {
  if (v >= 0.f) atomicMax((int*)addr, __float_as_int(v));
  else          atomicMin((unsigned int*)addr, __float_as_uint(v));
}

__global__ void att_logits_kernel(const int* __restrict__ src, const int* __restrict__ dst,
                                  const float* __restrict__ a_s, const float* __restrict__ a_d,
                                  float* __restrict__ evals, float* __restrict__ mmax) {
  int idx = blockIdx.x * blockDim.x + threadIdx.x;
  if (idx >= SI) return;
  int s, d;
  if (idx < N_EDGES) { s = src[idx]; d = dst[idx]; }
  else               { s = d = idx - N_EDGES; }
  float v = a_s[s] + a_d[d];
  v = v > 0.f ? v : 0.2f * v;  // leaky_relu(0.2)
  evals[idx] = v;
  atomicMaxF(&mmax[d], v);
}

__global__ void att_exp_kernel(const int* __restrict__ src, const int* __restrict__ dst,
                               const float* __restrict__ evals, const float* __restrict__ mmax,
                               float* __restrict__ exv, float* __restrict__ denom) {
  int idx = blockIdx.x * blockDim.x + threadIdx.x;
  if (idx >= SI) return;
  int d = (idx < N_EDGES) ? dst[idx] : idx - N_EDGES;
  float e = expf(evals[idx] - mmax[d]);
  exv[idx] = e;
  atomicAdd(&denom[d], e);
}

__global__ void att_agg_kernel(const int* __restrict__ src, const int* __restrict__ dst,
                               const float* __restrict__ exv, const float* __restrict__ denom,
                               const float* __restrict__ hp, float* __restrict__ h2acc) {
  int idx = blockIdx.x;   // one block per edge
  int o = threadIdx.x;    // 128 features
  int s, d;
  if (idx < N_EDGES) { s = src[idx]; d = dst[idx]; }
  else               { s = d = idx - N_EDGES; }
  float alpha = exv[idx] / (denom[d] + 1e-16f);
  atomicAdd(&h2acc[(size_t)d * HID + o], alpha * hp[(size_t)s * HID + o]);
}

// ---- relu(+gat_b) and mean-pool scatter ----
__global__ void pool_kernel(const float* __restrict__ h2acc, const float* __restrict__ gb,
                            const int* __restrict__ batch, float* __restrict__ pooled,
                            float* __restrict__ cnt) {
  int tid = blockIdx.x * blockDim.x + threadIdx.x;
  if (tid >= N_NODES * HID) return;
  int nd = tid >> 7, o = tid & 127;
  float v = h2acc[tid] + gb[o];
  v = v > 0.f ? v : 0.f;
  int b = batch[nd];
  atomicAdd(&pooled[b * HID + o], v);
  if (o == 0) atomicAdd(&cnt[b], 1.f);
}

// ---- z = relu((pooled/cnt)@fc1_w + fc1_b) ----
__global__ void fc1_kernel(const float* __restrict__ pooled, const float* __restrict__ cnt,
                           const float* __restrict__ w, const float* __restrict__ bias,
                           float* __restrict__ z) {
  int tid = blockIdx.x * blockDim.x + threadIdx.x;
  if (tid >= NGR * OUTF) return;
  int b = tid >> 6, o = tid & 63;
  float c = cnt[b];
  float den = c > 1.f ? c : 1.f;
  float acc = bias[o];
  for (int j = 0; j < HID; ++j) acc += (pooled[b * HID + j] / den) * w[j * OUTF + o];
  z[tid] = acc > 0.f ? acc : 0.f;
}

// ---- sigmoid(|z1-z2|@fc2_w + fc2_b) ----
__global__ void final_kernel(const float* __restrict__ z1, const float* __restrict__ z2,
                             const float* __restrict__ fw, const float* __restrict__ fb,
                             float* __restrict__ out) {
  int b = threadIdx.x;
  if (b >= NGR) return;
  float acc = fb[0];
  for (int j = 0; j < OUTF; ++j) acc += fabsf(z1[b * OUTF + j] - z2[b * OUTF + j]) * fw[j];
  out[b] = 1.f / (1.f + expf(-acc));
}

extern "C" void kernel_launch(void* const* d_in, const int* in_sizes, int n_in,
                              void* d_out, int out_size, void* d_ws, size_t ws_size,
                              hipStream_t stream) {
  const float* x[2]     = {(const float*)d_in[0], (const float*)d_in[4]};
  const int*   ei[2]    = {(const int*)d_in[1],   (const int*)d_in[5]};
  const float* ea[2]    = {(const float*)d_in[2], (const float*)d_in[6]};
  const int*   batch[2] = {(const int*)d_in[3],   (const int*)d_in[7]};
  const float* e_w1 = (const float*)d_in[8];
  const float* e_b1 = (const float*)d_in[9];
  const float* e_w2 = (const float*)d_in[10];
  const float* e_b2 = (const float*)d_in[11];
  const float* e_w3 = (const float*)d_in[12];
  const float* e_b3 = (const float*)d_in[13];
  const float* root_w  = (const float*)d_in[14];
  const float* conv1_b = (const float*)d_in[15];
  const float* gat_w   = (const float*)d_in[16];
  const float* att_src = (const float*)d_in[17];
  const float* att_dst = (const float*)d_in[18];
  const float* gat_b   = (const float*)d_in[19];
  const float* fc1_w = (const float*)d_in[20];
  const float* fc1_b = (const float*)d_in[21];
  const float* fc2_w = (const float*)d_in[22];
  const float* fc2_b = (const float*)d_in[23];

  float* wsf = (float*)d_ws;
  // bf16 pack area
  __bf16* w3p   = (__bf16*)wsf;                       // 524288 bf16 (262144 f)
  __bf16* w1p   = (__bf16*)(wsf + 262144);            // 4096 bf16
  __bf16* w2p   = (__bf16*)(wsf + 262144 + 2048);     // 8192 bf16
  __bf16* rootp = (__bf16*)(wsf + 262144 + 6144);     // 8192 bf16
  __bf16* gatp  = (__bf16*)(wsf + 262144 + 10240);    // 16384 bf16
  const size_t PACK_F = 262144 + 32768;
  const size_t PG = 1048576 + 3 * 524288 + 4 * 4096 + 2 * 20480 + 524288 + 4096 + 32 + 2048;

  pack_w3_kernel<<<2048, 256, 0, stream>>>(e_w3, w3p);
  pack_b_kernel<<<16, 256, 0, stream>>>(e_w1, w1p, F_EDGE, HID, 1, 8);
  pack_b_kernel<<<32, 256, 0, stream>>>(e_w2, w2p, HID, 64, 4, 4);
  pack_b_kernel<<<32, 256, 0, stream>>>(root_w, rootp, F_IN, HID, 2, 8);
  pack_b_kernel<<<64, 256, 0, stream>>>(gat_w, gatp, HID, HID, 4, 8);

  float* zg[2];
  for (int g = 0; g < 2; ++g) {
    float* base = wsf + PACK_F + (size_t)g * PG;
    float* h2e    = base;                  // E*64
    float* agg    = h2e + 1048576;         // N*128
    float* h1     = agg + 524288;          // N*128
    float* hp     = h1 + 524288;           // N*128
    float* a_s    = hp + 524288;           // N
    float* a_d    = a_s + 4096;            // N
    float* mmax   = a_d + 4096;            // N
    float* denom  = mmax + 4096;           // N
    float* evals  = denom + 4096;          // SI
    float* exv    = evals + 20480;         // SI
    float* h2acc  = exv + 20480;           // N*128
    float* pooled = h2acc + 524288;        // 32*128
    float* cnt    = pooled + 4096;         // 32
    float* z      = cnt + 32;              // 32*64
    zg[g] = z;

    const int* srcp = ei[g];
    const int* dstp = ei[g] + N_EDGES;

    fill_kernel<<<(N_NODES * HID + 255) / 256, 256, 0, stream>>>(agg, 0.f, N_NODES * HID);
    fill_kernel<<<(N_NODES * HID + 255) / 256, 256, 0, stream>>>(h2acc, 0.f, N_NODES * HID);
    fill_kernel<<<(N_NODES + 255) / 256, 256, 0, stream>>>(mmax, -INFINITY, N_NODES);
    fill_kernel<<<(N_NODES + 255) / 256, 256, 0, stream>>>(denom, 0.f, N_NODES);
    fill_kernel<<<(NGR * HID + 255) / 256, 256, 0, stream>>>(pooled, 0.f, NGR * HID);
    fill_kernel<<<1, 256, 0, stream>>>(cnt, 0.f, NGR);

    edge_mlp_kernel<<<N_EDGES / 16, 32, 0, stream>>>(ea[g], w1p, e_b1, w2p, e_b2, h2e);
    nnconv_kernel<<<N_EDGES / 16, 32, 0, stream>>>(x[g], srcp, dstp, h2e, w3p, e_b3, agg);
    conv_root_kernel<<<N_NODES / 16, 32, 0, stream>>>(x[g], rootp, conv1_b, agg, h1);
    gat_proj_kernel<<<N_NODES / 16, 32, 0, stream>>>(h1, gatp, att_src, att_dst, hp, a_s, a_d);
    att_logits_kernel<<<(SI + 255) / 256, 256, 0, stream>>>(srcp, dstp, a_s, a_d, evals, mmax);
    att_exp_kernel<<<(SI + 255) / 256, 256, 0, stream>>>(srcp, dstp, evals, mmax, exv, denom);
    att_agg_kernel<<<SI, 128, 0, stream>>>(srcp, dstp, exv, denom, hp, h2acc);
    pool_kernel<<<(N_NODES * HID + 255) / 256, 256, 0, stream>>>(h2acc, gat_b, batch[g], pooled, cnt);
    fc1_kernel<<<(NGR * OUTF + 255) / 256, 256, 0, stream>>>(pooled, cnt, fc1_w, fc1_b, z);
  }

  final_kernel<<<1, 32, 0, stream>>>(zg[0], zg[1], fc2_w, fc2_b, (float*)d_out);
}
